// LSTMDecoder_18107582120287
// MI455X (gfx1250) — compile-verified, matrix operand order checked by reading, not run on hardware
//
#include <hip/hip_runtime.h>

typedef __bf16 bf16_t;
typedef bf16_t v16bf __attribute__((ext_vector_type(16)));
typedef bf16_t v8bf  __attribute__((ext_vector_type(8)));
typedef float  v8f   __attribute__((ext_vector_type(8)));

#define B_   256
#define S_   141
#define V_   1000
#define VP_  1008   // V padded to multiple of 16 (Wx / embedding pad)
#define VB_  1024   // V padded to multiple of 64 (out_W pad for N-blocking)
#define E_   256
#define H_   512
#define G_   2048   // 4*H

// ---- WMMA fragment load -----------------------------------------------------
// 16-bit A/B operand layout (CDNA5 ISA 7.12.2): lane L holds row (L&15),
// elems 0..7  = k0 + (L>>4)*8 + 0..7
// elems 8..15 = k0 + 16 + (L>>4)*8 + 0..7
// => two contiguous 16B loads from a row-major [rows, ld] bf16 matrix.
__device__ __forceinline__ v16bf load_frag(const bf16_t* __restrict__ base,
                                           int ld, int row, int k0, int hi) {
    const bf16_t* p = base + (size_t)row * ld + k0 + hi * 8;
    v8bf lo = *(const v8bf*)(p);
    v8bf hh = *(const v8bf*)(p + 16);
    return __builtin_shufflevector(lo, hh, 0,1,2,3,4,5,6,7,8,9,10,11,12,13,14,15);
}

__device__ __forceinline__ v8f wmma_bf16(v16bf a, v16bf b, v8f c) {
    // (neg_a, A, neg_b, B, c_mod, C, reuse_a, reuse_b)
    return __builtin_amdgcn_wmma_f32_16x16x32_bf16(false, a, false, b,
                                                   (short)0, c, false, false);
}

__device__ __forceinline__ float sigmoidf_(float x) {
    return 1.0f / (1.0f + __expf(-x));
}

// ---- one-time conversion / init --------------------------------------------
__global__ void init_ws_kernel(
    const float* __restrict__ emb_src, const float* __restrict__ wih_src,
    const float* __restrict__ whh_src, const float* __restrict__ outw_src,
    const float* __restrict__ enc_src, const float* __restrict__ bih,
    const float* __restrict__ bhh,
    bf16_t* __restrict__ emb, bf16_t* __restrict__ wih,
    bf16_t* __restrict__ whh, bf16_t* __restrict__ outw,
    bf16_t* __restrict__ enc, float* __restrict__ bsum,
    bf16_t* __restrict__ h0, float* __restrict__ c0)
{
    int tid = blockIdx.x * blockDim.x + threadIdx.x;
    int stride = gridDim.x * blockDim.x;
    for (int i = tid; i < VP_ * E_; i += stride) {
        int r = i / E_;
        emb[i] = (r < V_) ? (bf16_t)emb_src[(size_t)r * E_ + (i % E_)] : (bf16_t)0.0f;
    }
    for (int i = tid; i < G_ * 2 * E_; i += stride) wih[i] = (bf16_t)wih_src[i];
    for (int i = tid; i < G_ * H_;     i += stride) whh[i] = (bf16_t)whh_src[i];
    for (int i = tid; i < VB_ * H_;    i += stride) {
        int r = i / H_;
        outw[i] = (r < V_) ? (bf16_t)outw_src[(size_t)r * H_ + (i % H_)] : (bf16_t)0.0f;
    }
    for (int i = tid; i < B_ * E_; i += stride) enc[i] = (bf16_t)enc_src[i];
    for (int i = tid; i < G_;      i += stride) bsum[i] = bih[i] + bhh[i];
    for (int i = tid; i < B_ * H_; i += stride) h0[i] = (bf16_t)0.0f;
    for (int i = tid; i < B_ * H_; i += stride) c0[i] = 0.0f;
}

// ---- generic C = A @ B^T (+bias per column), bf16 in / f32 out --------------
// A: [Mtiles*16, K] row-major bf16 ; Bm rows are output columns: [Ntiles*16, K]
__global__ __launch_bounds__(128) void gemm_nt_kernel(
    const bf16_t* __restrict__ A, int lda,
    const bf16_t* __restrict__ Bm, int ldb,
    const float* __restrict__ bias,
    float* __restrict__ C, int ldc,
    int Mtiles, int Ntiles, int K)
{
    int wave = blockIdx.x * (blockDim.x >> 5) + (threadIdx.x >> 5);
    if (wave >= Mtiles * Ntiles) return;              // wave-uniform exit
    int mt = wave / Ntiles, nt = wave % Ntiles;
    int lane = threadIdx.x & 31;
    int l15 = lane & 15, hi = lane >> 4;
    int m0 = mt * 16, n0 = nt * 16;
    float bv = bias ? bias[n0 + l15] : 0.0f;
    v8f acc;
#pragma unroll
    for (int r = 0; r < 8; ++r) acc[r] = bv;
    for (int k0 = 0; k0 < K; k0 += 32) {
        v16bf a = load_frag(A,  lda, m0 + l15, k0, hi);
        v16bf b = load_frag(Bm, ldb, n0 + l15, k0, hi);
        acc = wmma_bf16(a, b, acc);
    }
    // C/D layout: lane(0..15) -> n=lane, m=r ; lane(16..31) -> n=lane-16, m=8+r
#pragma unroll
    for (int r = 0; r < 8; ++r)
        C[(size_t)(m0 + r + 8 * hi) * ldc + n0 + l15] = acc[r];
}

// ---- one fused LSTM step: gates GEMM (4 gate tiles/wave) + pointwise --------
__global__ __launch_bounds__(128) void lstm_step_kernel(
    const bf16_t* __restrict__ h_in,       // [B,H] bf16 (t-1 hidden)
    const bf16_t* __restrict__ whh,        // [4H,H] bf16 rows = gate outputs
    const float*  __restrict__ enc_gates,  // [B,4H] f32 (enc proj + biases)
    const float*  __restrict__ Wx,         // [VP,4H] f32 (embedding proj)
    const int*    __restrict__ tseq,       // [B,S] tokens
    int t,
    float*  __restrict__ c,                // [B,H] f32, in-place
    bf16_t* __restrict__ h_out)            // [B,H] bf16 = hs slice t
{
    int wave = blockIdx.x * (blockDim.x >> 5) + (threadIdx.x >> 5); // 0..511
    int bt = wave >> 5;        // B/16 = 16 tiles
    int ht = wave & 31;        // H/16 = 32 tiles
    int lane = threadIdx.x & 31;
    int l15 = lane & 15, hi = lane >> 4;
    int b0 = bt * 16, h0 = ht * 16;

    int tok[8];
#pragma unroll
    for (int r = 0; r < 8; ++r) tok[r] = tseq[(b0 + r + 8 * hi) * S_ + t];

    v8f acc[4];
#pragma unroll
    for (int q = 0; q < 4; ++q) {
        int g = q * H_ + h0 + l15;
#pragma unroll
        for (int r = 0; r < 8; ++r) {
            int bb = b0 + r + 8 * hi;
            acc[q][r] = enc_gates[(size_t)bb * G_ + g] + Wx[(size_t)tok[r] * G_ + g];
        }
    }
    for (int k0 = 0; k0 < H_; k0 += 32) {
        v16bf a = load_frag(h_in, H_, b0 + l15, k0, hi);   // shared by 4 gates
#pragma unroll
        for (int q = 0; q < 4; ++q) {
            v16bf b = load_frag(whh, H_, q * H_ + h0 + l15, k0, hi);
            acc[q] = wmma_bf16(a, b, acc[q]);
        }
    }
#pragma unroll
    for (int r = 0; r < 8; ++r) {
        int bb  = b0 + r + 8 * hi;
        int idx = bb * H_ + h0 + l15;
        float ig = sigmoidf_(acc[0][r]);
        float fg = sigmoidf_(acc[1][r]);
        float gg = tanhf(acc[2][r]);
        float og = sigmoidf_(acc[3][r]);
        float cn = fg * c[idx] + ig * gg;
        c[idx] = cn;
        h_out[idx] = (bf16_t)(og * tanhf(cn));
    }
}

// ---- logits[b,s,v] = hs[s,b,:] @ out_W[v,:] + out_b[v] ----------------------
// Register-blocked: each wave computes a 32(M) x 64(N) block = 2x4 WMMA tiles.
// Per k-step: 2 A-frags + 4 B-frags (12 b128) feed 8 WMMAs -> ~21 FLOP/byte.
__global__ __launch_bounds__(128) void proj_kernel(
    const bf16_t* __restrict__ hs,     // [S,B,H] bf16
    const bf16_t* __restrict__ outw,   // [VB,H] bf16 (zero padded)
    const float*  __restrict__ outb,   // [V]
    float* __restrict__ out)           // [B,S,V]
{
    const int MB = B_ / 32;    // 8  m-blocks
    const int NB = VB_ / 64;   // 16 n-blocks
    int wave = blockIdx.x * (blockDim.x >> 5) + (threadIdx.x >> 5);
    if (wave >= S_ * MB * NB) return;          // wave-uniform exit
    int nb = wave % NB;
    int rest = wave / NB;
    int mb = rest % MB;
    int s  = rest / MB;
    int lane = threadIdx.x & 31;
    int l15 = lane & 15, hi = lane >> 4;
    int b0 = mb * 32, v0 = nb * 64;

    v8f acc[2][4];
#pragma unroll
    for (int j = 0; j < 4; ++j) {
        int v = v0 + j * 16 + l15;
        float bias = (v < V_) ? outb[v] : 0.0f;
#pragma unroll
        for (int r = 0; r < 8; ++r) { acc[0][j][r] = bias; acc[1][j][r] = bias; }
    }

    const bf16_t* Abase = hs + (size_t)s * B_ * H_;
    for (int k0 = 0; k0 < H_; k0 += 32) {
        v16bf a0 = load_frag(Abase, H_, b0 + l15,      k0, hi);
        v16bf a1 = load_frag(Abase, H_, b0 + 16 + l15, k0, hi);
#pragma unroll
        for (int j = 0; j < 4; ++j) {
            v16bf b = load_frag(outw, H_, v0 + j * 16 + l15, k0, hi);
            acc[0][j] = wmma_bf16(a0, b, acc[0][j]);
            acc[1][j] = wmma_bf16(a1, b, acc[1][j]);
        }
    }
#pragma unroll
    for (int i = 0; i < 2; ++i) {
#pragma unroll
        for (int j = 0; j < 4; ++j) {
            int v = v0 + j * 16 + l15;
            if (v < V_) {
#pragma unroll
                for (int r = 0; r < 8; ++r)
                    out[(size_t)(b0 + i * 16 + r + 8 * hi) * (S_ * V_)
                        + (size_t)s * V_ + v] = acc[i][j][r];
            }
        }
    }
}

// ---- host side --------------------------------------------------------------
extern "C" void kernel_launch(void* const* d_in, const int* in_sizes, int n_in,
                              void* d_out, int out_size, void* d_ws, size_t ws_size,
                              hipStream_t stream) {
    (void)in_sizes; (void)n_in; (void)out_size; (void)ws_size;
    const float* enc  = (const float*)d_in[0];
    const int*   tseq = (const int*)  d_in[1];
    const float* emb  = (const float*)d_in[2];
    const float* wih  = (const float*)d_in[3];
    const float* whh  = (const float*)d_in[4];
    const float* bih  = (const float*)d_in[5];
    const float* bhh  = (const float*)d_in[6];
    // d_in[7..9] = attn_W / attn_b / v_w : numerically dead (softmax over 1)
    const float* outW = (const float*)d_in[10];
    const float* outb = (const float*)d_in[11];
    float* out = (float*)d_out;

    char* ws = (char*)d_ws;                          // all offsets 256B aligned
    bf16_t* emb_bf  = (bf16_t*)(ws + 0);             // 1008*256*2   = 516096
    bf16_t* wih_bf  = (bf16_t*)(ws + 516096);        // 2048*1024*2  = 4194304
    bf16_t* whh_bf  = (bf16_t*)(ws + 4710400);       // 2048*512*2   = 2097152
    bf16_t* outw_bf = (bf16_t*)(ws + 6807552);       // 1024*512*2   = 1048576
    bf16_t* enc_bf  = (bf16_t*)(ws + 7856128);       // 256*256*2    = 131072
    float*  bsum    = (float*) (ws + 7987200);       // 2048*4       = 8192
    bf16_t* h0      = (bf16_t*)(ws + 7995392);       // 256*512*2    = 262144
    float*  cbuf    = (float*) (ws + 8257536);       // 256*512*4    = 524288
    float*  Wx      = (float*) (ws + 8781824);       // 1008*2048*4  = 8257536
    float*  eg      = (float*) (ws + 17039360);      // 256*2048*4   = 2097152
    bf16_t* hs      = (bf16_t*)(ws + 19136512);      // 141*256*512*2 -> ~56MB end

    init_ws_kernel<<<2048, 256, 0, stream>>>(emb, wih, whh, outW, enc, bih, bhh,
        emb_bf, wih_bf, whh_bf, outw_bf, enc_bf, bsum, h0, cbuf);

    {   // Wx = embedding @ W_ih[:, :E].T         [VP, 4H]
        int Mt = VP_ / 16, Nt = G_ / 16, waves = Mt * Nt;
        gemm_nt_kernel<<<(waves + 3) / 4, 128, 0, stream>>>(
            emb_bf, E_, wih_bf, 2 * E_, nullptr, Wx, G_, Mt, Nt, E_);
    }
    {   // enc_gates = enc @ W_ih[:, E:].T + (b_ih + b_hh)   [B, 4H]
        int Mt = B_ / 16, Nt = G_ / 16, waves = Mt * Nt;
        gemm_nt_kernel<<<(waves + 3) / 4, 128, 0, stream>>>(
            enc_bf, E_, wih_bf + E_, 2 * E_, bsum, eg, G_, Mt, Nt, E_);
    }
    for (int t = 0; t < S_; ++t) {
        const bf16_t* hin = (t == 0) ? h0 : (hs + (size_t)(t - 1) * B_ * H_);
        bf16_t* hout = hs + (size_t)t * B_ * H_;
        lstm_step_kernel<<<128, 128, 0, stream>>>(hin, whh_bf, eg, Wx, tseq, t,
                                                  cbuf, hout);
    }
    {
        int waves = S_ * (B_ / 32) * (VB_ / 64);     // 18048
        proj_kernel<<<(waves + 3) / 4, 128, 0, stream>>>(hs, outw_bf, outb, out);
    }
}
